// CnnSelfAttention_52802327937692
// MI455X (gfx1250) — compile-verified, matrix operand order checked by reading, not run on hardware
//
#include <hip/hip_runtime.h>
#include <hip/hip_bf16.h>
#include <math.h>

typedef __attribute__((ext_vector_type(16))) __bf16 v16bf;
typedef __attribute__((ext_vector_type(8)))  float  v8f;
typedef __attribute__((ext_vector_type(4)))  float  v4f;

#define B_   2
#define T_   64
#define CIN  16
#define H_   64
#define W_   64
#define OC_  64
#define NH_  8
#define HC_  8
#define HW_  4096
#define D_   32768
#define BT_  128
#define K1   144
#define K1CH 5     // ceil(144/32)
#define K2   576
#define K2CH 18    // 576/32
#define NSLICE 16
#define SLICE_LEN 2048   // D_/NSLICE

union FragU { v16bf v; v4f f4[2]; __bf16 e[16]; };

__device__ inline v16bf ld_frag(const __bf16* p0, const __bf16* p1) {
  FragU u;
  u.f4[0] = *(const v4f*)p0;
  u.f4[1] = *(const v4f*)p1;
  return u.v;
}

__device__ inline v8f wmma_bf16(v16bf a, v16bf b, v8f c) {
  // 8 args: (neg_a, A, neg_b, B, c_mod, C, reuse_a, reuse_b)
  return __builtin_amdgcn_wmma_f32_16x16x32_bf16(false, a, false, b, (short)0, c,
                                                 false, false);
}

// ---------------------------------------------------------------------------
// Kernel 1: q/k/v 3x3 convs as implicit GEMM (M=BT*HW, N=OC=64, K=144->160).
// Shared im2col tile (A) reused for all three weight sets. Outputs written
// bf16 directly in attention layout [B, nh, T, hc*H*W].
// ---------------------------------------------------------------------------
__global__ __launch_bounds__(128)
void qkv_conv_kernel(const float* __restrict__ x,
                     const float* __restrict__ wq, const float* __restrict__ bq,
                     const float* __restrict__ wk, const float* __restrict__ bk,
                     const float* __restrict__ wv, const float* __restrict__ bv,
                     __bf16* __restrict__ q, __bf16* __restrict__ k,
                     __bf16* __restrict__ v)
{
  __shared__ __bf16 As[64][40];        // 64 rows x 32 K-chunk (pad 40: 80B, 16B-aligned)
  __shared__ __bf16 Bs[3][64][32];     // per-conv weight chunk, N-major / K-contig
  const int tid = threadIdx.x;
  const int wid = tid >> 5, lane = tid & 31;
  const int g = lane >> 4, ln = lane & 15;
  const int m0 = blockIdx.x * 64;      // 64 consecutive output positions
  const int bt = m0 >> 12;             // HW=4096 positions per image
  const int posbase = m0 & 4095;

  __builtin_prefetch(wq, 0, 3);
  __builtin_prefetch(wk, 0, 3);
  __builtin_prefetch(wv, 0, 3);

  v8f acc[3][4] = {};                  // 3 convs x 4 N-tiles of 16

  for (int kc = 0; kc < K1CH; ++kc) {
    __syncthreads();
    // build im2col chunk (zero-padded to K=160)
    for (int idx = tid; idx < 64 * 32; idx += 128) {
      int r = idx >> 5, kk = idx & 31;
      int K = kc * 32 + kk;
      float val = 0.0f;
      if (K < K1) {
        int c = K / 9, p = K % 9;
        int dh = p / 3 - 1, dw = p % 3 - 1;
        int pos = posbase + r;
        int h = pos >> 6, w = pos & 63;
        int hh = h + dh, ww = w + dw;
        if (hh >= 0 && hh < H_ && ww >= 0 && ww < W_)
          val = x[((size_t)(bt * CIN + c) << 12) + (hh << 6) + ww];
      }
      As[r][kk] = (__bf16)val;
    }
    // weight chunks for all three convs
    for (int idx = tid; idx < 3 * 64 * 32; idx += 128) {
      int cv = idx >> 11;
      int rem = idx & 2047;
      int n = rem >> 5, kk = rem & 31;
      int K = kc * 32 + kk;
      float val = 0.0f;
      if (K < K1) {
        const float* w = (cv == 0) ? wq : (cv == 1) ? wk : wv;
        val = w[n * K1 + K];
      }
      Bs[cv][n][kk] = (__bf16)val;
    }
    __syncthreads();

    v16bf a = ld_frag(&As[wid * 16 + ln][8 * g], &As[wid * 16 + ln][16 + 8 * g]);
    for (int cv = 0; cv < 3; ++cv)
      for (int nt = 0; nt < 4; ++nt) {
        v16bf b = ld_frag(&Bs[cv][nt * 16 + ln][8 * g],
                          &Bs[cv][nt * 16 + ln][16 + 8 * g]);
        acc[cv][nt] = wmma_bf16(a, b, acc[cv][nt]);
      }
  }

  const int b_ = bt / T_, t_ = bt % T_;
  for (int cv = 0; cv < 3; ++cv) {
    __bf16* dst = (cv == 0) ? q : (cv == 1) ? k : v;
    const float* bias = (cv == 0) ? bq : (cv == 1) ? bk : bv;
    for (int nt = 0; nt < 4; ++nt) {
      int oc = nt * 16 + ln;
      float bb = bias[oc];
      int head = oc >> 3, hcx = oc & 7;
      size_t base = (((size_t)(b_ * NH_ + head) * T_ + t_) * D_) + ((size_t)hcx << 12);
      for (int r = 0; r < 8; ++r) {
        int mloc = wid * 16 + r + 8 * g;
        dst[base + posbase + mloc] = (__bf16)(acc[cv][nt][r] + bb);
      }
    }
  }
}

// ---------------------------------------------------------------------------
// Kernel 2a: zero attention accumulator.
// ---------------------------------------------------------------------------
__global__ void zero_att_kernel(float* __restrict__ att)
{
  int i = blockIdx.x * blockDim.x + threadIdx.x;
#pragma unroll
  for (int j = 0; j < 4; ++j) att[i * 4 + j] = 0.0f;   // 64*256*4 = 65536
}

// ---------------------------------------------------------------------------
// Kernel 2b: att += q @ k^T  (split-K over d=32768, 16 slices of 2048).
// One wave per (head, 16x16 tile, slice); bf16 WMMA, f32 atomic accumulate.
// ---------------------------------------------------------------------------
__global__ __launch_bounds__(32)
void qk_kernel(const __bf16* __restrict__ q, const __bf16* __restrict__ k,
               float* __restrict__ att)
{
  const int bid = blockIdx.x;
  const int slice = bid & (NSLICE - 1);
  const int tile = (bid >> 4) & 15;
  const int head = bid >> 8;
  const int mt = tile >> 2, ntv = tile & 3;
  const int lane = threadIdx.x;
  const int g = lane >> 4, ln = lane & 15;

  const __bf16* qrow = q + (size_t)(head * T_ + mt * 16 + ln) * D_ + slice * SLICE_LEN;
  const __bf16* krow = k + (size_t)(head * T_ + ntv * 16 + ln) * D_ + slice * SLICE_LEN;

  v8f acc = {};
  for (int c = 0; c < SLICE_LEN / 32; ++c) {
    const __bf16* qp = qrow + c * 32;
    const __bf16* kp = krow + c * 32;
    v16bf a = ld_frag(qp + 8 * g, qp + 16 + 8 * g);
    v16bf b = ld_frag(kp + 8 * g, kp + 16 + 8 * g);
    acc = wmma_bf16(a, b, acc);
  }
#pragma unroll
  for (int r = 0; r < 8; ++r) {
    int qt = mt * 16 + r + 8 * g;
    int kt = ntv * 16 + ln;
    atomicAdd(&att[(size_t)head * 4096 + qt * 64 + kt], acc[r]);
  }
}

// ---------------------------------------------------------------------------
// Kernel 3: causal softmax over 64 keys per row; emit bf16 probabilities.
// ---------------------------------------------------------------------------
__global__ __launch_bounds__(32)
void softmax_kernel(const float* __restrict__ att, __bf16* __restrict__ p)
{
  const int row = blockIdx.x;        // head*64 + qt
  const int qt = row & 63;
  const int lane = threadIdx.x;
  const float scale = 0.005524271728019903f;   // 1/sqrt(32768)
  const float* ar = att + (size_t)row * 64;
  int k0 = lane, k1 = lane + 32;
  float x0 = (k0 <= qt) ? ar[k0] * scale : -3.0e38f;
  float x1 = (k1 <= qt) ? ar[k1] * scale : -3.0e38f;
  float m = fmaxf(x0, x1);
  for (int off = 16; off > 0; off >>= 1) m = fmaxf(m, __shfl_xor(m, off));
  float e0 = (k0 <= qt) ? __expf(x0 - m) : 0.0f;
  float e1 = (k1 <= qt) ? __expf(x1 - m) : 0.0f;
  float s = e0 + e1;
  for (int off = 16; off > 0; off >>= 1) s += __shfl_xor(s, off);
  float inv = 1.0f / s;
  __bf16* pr = p + (size_t)row * 64;
  pr[k0] = (__bf16)(e0 * inv);
  pr[k1] = (__bf16)(e1 * inv);
}

// ---------------------------------------------------------------------------
// Kernel 4: y = P @ V per head, tiled over d. P tile and V block staged into
// LDS with CDNA5 async global->LDS copies; bf16 WMMA; y stored bf16 already
// permuted back to [BT, OC, H, W].
// ---------------------------------------------------------------------------
__global__ __launch_bounds__(128)
void pv_kernel(const __bf16* __restrict__ p, const __bf16* __restrict__ v,
               __bf16* __restrict__ y)
{
  __shared__ __bf16 Pg[64][64];      // probabilities (qt x kt)
  __shared__ __bf16 Vs[64][72];      // V block (kt x 64 dd), pad->144B stride
  const int tid = threadIdx.x;
  const int head = blockIdx.x >> 9;  // 512 d-chunks of 64
  const int d0 = (blockIdx.x & 511) << 6;

  {
    const __bf16* psrc = p + (size_t)head * 4096;
#pragma unroll
    for (int i = 0; i < 4; ++i) {   // 512 x 16B units for P
      int u = tid + i * 128;
      unsigned lds = (unsigned)(unsigned long long)(&Pg[0][0] + u * 8);
      unsigned long long ga = (unsigned long long)(psrc + u * 8);
      asm volatile("global_load_async_to_lds_b128 %0, %1, off"
                   :: "v"(lds), "v"(ga) : "memory");
    }
    const __bf16* vsrc = v + (size_t)head * T_ * D_ + d0;
#pragma unroll
    for (int i = 0; i < 4; ++i) {   // 64 rows x 8 x 16B units for V
      int u = tid + i * 128;
      int r = u >> 3, cu = u & 7;
      unsigned lds = (unsigned)(unsigned long long)(&Vs[r][cu * 8]);
      unsigned long long ga = (unsigned long long)(vsrc + (size_t)r * D_ + cu * 8);
      asm volatile("global_load_async_to_lds_b128 %0, %1, off"
                   :: "v"(lds), "v"(ga) : "memory");
    }
    asm volatile("s_wait_asynccnt 0" ::: "memory");
  }
  __syncthreads();

  const int wid = tid >> 5, lane = tid & 31, g = lane >> 4, ln = lane & 15;
  v8f acc[4] = {};
  for (int kc = 0; kc < 2; ++kc) {
    v16bf a = ld_frag(&Pg[wid * 16 + ln][kc * 32 + 8 * g],
                      &Pg[wid * 16 + ln][kc * 32 + 16 + 8 * g]);
    for (int nt = 0; nt < 4; ++nt) {
      FragU u;
#pragma unroll
      for (int j = 0; j < 16; ++j) {
        int kk = kc * 32 + ((j < 8) ? j : 8 + j) + 8 * g;
        u.e[j] = Vs[kk][nt * 16 + ln];
      }
      acc[nt] = wmma_bf16(a, u.v, acc[nt]);
    }
  }

  const int b_ = head >> 3, n_ = head & 7;
  for (int nt = 0; nt < 4; ++nt)
    for (int r = 0; r < 8; ++r) {
      int qt = wid * 16 + r + 8 * g;
      int dd = d0 + nt * 16 + ln;
      int hcx = dd >> 12, pos = dd & 4095;
      int btq = b_ * T_ + qt;
      int oc = n_ * HC_ + hcx;
      y[(((size_t)btq * OC_ + oc) << 12) + pos] = (__bf16)acc[nt][r];
    }
}

// ---------------------------------------------------------------------------
// Kernel 5: output 3x3 conv (64->64) as implicit GEMM, K=576=18x32 exactly.
// Reads bf16 y, writes f32 result (+bias) in [B,T,OC,H,W] layout.
// ---------------------------------------------------------------------------
__global__ __launch_bounds__(128)
void oconv_kernel(const __bf16* __restrict__ y, const float* __restrict__ wo,
                  const float* __restrict__ bo, float* __restrict__ out)
{
  __shared__ __bf16 As[64][40];
  __shared__ __bf16 Bs[64][32];
  const int tid = threadIdx.x, wid = tid >> 5, lane = tid & 31;
  const int g = lane >> 4, ln = lane & 15;
  const int m0 = blockIdx.x * 64;
  const int bt = m0 >> 12;
  const int posbase = m0 & 4095;

  __builtin_prefetch(wo, 0, 3);

  v8f acc[4] = {};
  for (int kc = 0; kc < K2CH; ++kc) {
    __syncthreads();
    for (int idx = tid; idx < 64 * 32; idx += 128) {
      int r = idx >> 5, kk = idx & 31;
      int K = kc * 32 + kk;
      int c = K / 9, pp = K % 9;
      int dh = pp / 3 - 1, dw = pp % 3 - 1;
      int pos = posbase + r;
      int h = pos >> 6, w = pos & 63;
      int hh = h + dh, ww = w + dw;
      __bf16 val = (__bf16)0.0f;
      if (hh >= 0 && hh < H_ && ww >= 0 && ww < W_)
        val = y[(((size_t)bt * OC_ + c) << 12) + (hh << 6) + ww];
      As[r][kk] = val;
    }
    for (int idx = tid; idx < 64 * 32; idx += 128) {
      int n = idx >> 5, kk = idx & 31;
      Bs[n][kk] = (__bf16)wo[n * K2 + kc * 32 + kk];
    }
    __syncthreads();

    v16bf a = ld_frag(&As[wid * 16 + ln][8 * g], &As[wid * 16 + ln][16 + 8 * g]);
    for (int nt = 0; nt < 4; ++nt) {
      v16bf b = ld_frag(&Bs[nt * 16 + ln][8 * g], &Bs[nt * 16 + ln][16 + 8 * g]);
      acc[nt] = wmma_bf16(a, b, acc[nt]);
    }
  }

  for (int nt = 0; nt < 4; ++nt) {
    int oc = nt * 16 + ln;
    float bb = bo[oc];
    for (int r = 0; r < 8; ++r) {
      int mloc = wid * 16 + r + 8 * g;
      out[(((size_t)bt * OC_ + oc) << 12) + posbase + mloc] = acc[nt][r] + bb;
    }
  }
}

// ---------------------------------------------------------------------------
// Host-side launcher. Workspace layout (bytes):
//   q bf16 67MB | k bf16 67MB | v bf16 67MB | y bf16 67MB | att f32 256KB |
//   p bf16 128KB   => ~269 MB total.
// ---------------------------------------------------------------------------
extern "C" void kernel_launch(void* const* d_in, const int* in_sizes, int n_in,
                              void* d_out, int out_size, void* d_ws, size_t ws_size,
                              hipStream_t stream)
{
  const float* x  = (const float*)d_in[0];
  const float* wq = (const float*)d_in[1];
  const float* bq = (const float*)d_in[2];
  const float* wk = (const float*)d_in[3];
  const float* bk = (const float*)d_in[4];
  const float* wv = (const float*)d_in[5];
  const float* bv = (const float*)d_in[6];
  const float* wo = (const float*)d_in[7];
  const float* bo = (const float*)d_in[8];
  float* out = (float*)d_out;

  const size_t QN = (size_t)B_ * NH_ * T_ * D_;   // 33,554,432 elements
  __bf16* q = (__bf16*)d_ws;
  __bf16* k = q + QN;
  __bf16* v = k + QN;
  __bf16* y = v + QN;
  float* att = (float*)(y + QN);                  // B*nh*T*T = 65536 f32
  __bf16* p = (__bf16*)(att + (size_t)B_ * NH_ * T_ * T_);

  qkv_conv_kernel<<<8192, 128, 0, stream>>>(x, wq, bq, wk, bk, wv, bv, q, k, v);
  zero_att_kernel<<<64, 256, 0, stream>>>(att);
  qk_kernel<<<16 * 16 * NSLICE, 32, 0, stream>>>(q, k, att);
  softmax_kernel<<<B_ * NH_ * T_, 32, 0, stream>>>(att, p);
  pv_kernel<<<16 * 512, 128, 0, stream>>>(p, v, y);
  oconv_kernel<<<8192, 128, 0, stream>>>(y, wo, bo, out);
}
